// hybrid_Attention_25555055411274
// MI455X (gfx1250) — matrix operand order repaired and checked by reading, not compiled, not run on hardware
//
#include <hip/hip_runtime.h>
#include <hip/hip_bf16.h>

// ---------------------------------------------------------------------------
// MI455X (gfx1250) hybrid window + PVT attention, f16 WMMA everywhere GEMM-shaped.
//   B=2, X=Y=24, WS=8, d=256, d2=128, heads=8 (dh2=16), hg=4, DH=32
// Async global->LDS staging (GLOBAL_LOAD_ASYNC_TO_LDS_B128) for input tiles.
// ---------------------------------------------------------------------------

typedef __attribute__((ext_vector_type(16))) _Float16 v16h;
typedef __attribute__((ext_vector_type(8)))  float    v8f;

#define SCALE 0.17677669529663687f   // 1/sqrt(32)

__device__ __forceinline__ v8f zerov() { v8f z = {}; return z; }

__device__ __forceinline__ v8f wmma16(v16h a, v16h b, v8f c) {
  return __builtin_amdgcn_wmma_f32_16x16x32_f16(false, a, false, b, (short)0, c,
                                                false, false);
}

// Async copy of 16B global -> LDS (CDNA5 GLOBAL_LOAD_ASYNC_TO_LDS_B128, GV mode).
// LDS generic pointers carry the LDS byte offset in addr[31:0] (ISA 10.2).
__device__ __forceinline__ void async_ld_b128(void* lds_dst, const void* gsrc) {
  unsigned ldsoff = (unsigned)(unsigned long long)lds_dst;
  asm volatile("global_load_async_to_lds_b128 %0, %1, off"
               :: "v"(ldsoff), "v"(gsrc) : "memory");
}
__device__ __forceinline__ void wait_async0() {
  asm volatile("s_wait_asynccnt 0" ::: "memory");
}

// A fragment (16x32, f16) from row-major [M][K] tile base, ld in elements.
// ISA layout: lanes 0-15 -> M=lane, Kbase=0; lanes 16-31 -> M=lane-16, Kbase=8;
// VGPR 0-3 hold K=Kbase..Kbase+7 (2/VGPR), VGPR 4-7 hold K=16+Kbase..16+Kbase+7.
__device__ __forceinline__ v16h load_a_frag(const _Float16* A, int ld, int lane) {
  int m  = lane & 15;
  int kb = (lane & 16) ? 8 : 0;
  const _Float16* row = A + m * ld;
  v16h a;
#pragma unroll
  for (int e = 0; e < 16; ++e) {
    int v  = e >> 1;
    int kk = (v < 4) ? (kb + (v << 1) + (e & 1))
                     : (16 + kb + ((v - 4) << 1) + (e & 1));
    a[e] = row[kk];
  }
  return a;
}

// B fragment (32x16, f16) from row-major [K][N] tile base.
// lanes 0-15 -> N=lane, K=0..7,16..23; lanes 16-31 -> N=lane-16, K=8..15,24..31.
__device__ __forceinline__ v16h load_b_frag(const _Float16* B, int ld, int lane) {
  int n  = lane & 15;
  int kb = (lane & 16) ? 8 : 0;
  v16h b;
#pragma unroll
  for (int e = 0; e < 16; ++e) b[e] = B[(kb + (e & 7) + ((e >> 3) << 4)) * ld + n];
  return b;
}

// ---------------------------------------------------------------------------
// Prep: f32 -> f16 conversions (weights once; whole input x once)
// ---------------------------------------------------------------------------
__global__ void cvt_f16_kernel(const float* __restrict__ src,
                               _Float16* __restrict__ dst, int n) {
  int i = blockIdx.x * 256 + threadIdx.x;
  if (i < n) dst[i] = (_Float16)src[i];
}

// cu_w (O=128, I=128, 3, 3) -> cu16[tap][ci][co]
__global__ void cvt_cu_kernel(const float* __restrict__ cu_w,
                              _Float16* __restrict__ cu16) {
  int i = blockIdx.x * 256 + threadIdx.x;
  if (i < 9 * 128 * 128) {
    int tap = i / 16384, r = i % 16384, ci = r / 128, co = r % 128;
    cu16[i] = (_Float16)cu_w[(co * 128 + ci) * 9 + tap];
  }
}

// ---------------------------------------------------------------------------
// Kernel 1: full window-attention branch. One block per window (1152), 8 waves.
// LDS: sA 64x264 f16 | sXq 64x136 f16 | sQKV 64x392 f16 | sSim 8x64x68 f32
// ---------------------------------------------------------------------------
__global__ void __launch_bounds__(256)
win_attn_kernel(const _Float16* __restrict__ x16, const _Float16* __restrict__ q1w16,
                const float* __restrict__ q1_b, const _Float16* __restrict__ wqkv16,
                const _Float16* __restrict__ wout16,
                const float* __restrict__ rel_table, float* __restrict__ out) {
  extern __shared__ char smem[];
  _Float16* sA   = (_Float16*)smem;                                 // 33792 B
  _Float16* sXq  = (_Float16*)(smem + 33792);                       // 17408 B
  _Float16* sQKV = (_Float16*)(smem + 33792 + 17408);               // 50176 B
  float*    sSim = (float*)(smem + 33792 + 17408 + 50176);          // 139264 B

  const int tid = threadIdx.x, lane = tid & 31, w = tid >> 5;
  const int wi  = blockIdx.x;

  // warm L2->WGP path for the shared f16 weights (global_prefetch_b8)
  {
    const char* wp = (const char*)q1w16;   // q1w16|wqkv16|wout16 contiguous-ish
    __builtin_prefetch(wp + tid * 256, 0, 0);
    __builtin_prefetch((const char*)wqkv16 + tid * 384, 0, 0);
    __builtin_prefetch((const char*)wout16 + tid * 128, 0, 0);
  }

  // async stage window tile (64 x 256 f16) global -> LDS, 16B granules
  {
    const char* src = (const char*)(x16 + (size_t)wi * 64 * 256);
    for (int i = tid; i < 2048; i += 256) {        // 64 rows x 32 chunks
      int row = i >> 5, ch = i & 31;
      async_ld_b128((char*)sA + row * 528 + ch * 16, src + row * 512 + ch * 16);
    }
    wait_async0();
  }
  __syncthreads();

  // GEMM1: xq(64x128) = A @ q1_w + q1_b.  wave -> N tile.
  {
    int n0 = w * 16;
    float bias = q1_b[n0 + (lane & 15)];
    for (int mt = 0; mt < 4; ++mt) {
      v8f acc = zerov();
      for (int kt = 0; kt < 8; ++kt) {
        v16h a = load_a_frag(sA + mt * 16 * 264 + kt * 32, 264, lane);
        v16h b = load_b_frag(q1w16 + kt * 32 * 128 + n0, 128, lane);
        acc = wmma16(a, b, acc);
      }
      int hi = (lane >> 4) << 3;
#pragma unroll
      for (int r = 0; r < 8; ++r)
        sXq[(mt * 16 + hi + r) * 136 + n0 + (lane & 15)] = (_Float16)(acc[r] + bias);
    }
  }
  __syncthreads();

  // GEMM2: qkv(64x384) = xq @ w_qkv  (q part pre-scaled)
  for (int nt = w; nt < 24; nt += 8) {
    int n0 = nt * 16;
    float sc = (n0 < 128) ? SCALE : 1.0f;
    for (int mt = 0; mt < 4; ++mt) {
      v8f acc = zerov();
      for (int kt = 0; kt < 4; ++kt) {
        v16h a = load_a_frag(sXq + mt * 16 * 136 + kt * 32, 136, lane);
        v16h b = load_b_frag(wqkv16 + kt * 32 * 384 + n0, 384, lane);
        acc = wmma16(a, b, acc);
      }
      int hi = (lane >> 4) << 3;
#pragma unroll
      for (int r = 0; r < 8; ++r)
        sQKV[(mt * 16 + hi + r) * 392 + n0 + (lane & 15)] = (_Float16)(acc[r] * sc);
    }
  }
  __syncthreads();

  // sim(64x64) per head (wave == head), K=16 padded to 32
  {
    int qoff = w * 16, koff = 128 + w * 16;
    for (int mt = 0; mt < 4; ++mt)
      for (int nt = 0; nt < 4; ++nt) {
        v16h a, b;
        {
          int m = lane & 15, kb = (lane & 16) ? 8 : 0;
          const _Float16* row = sQKV + (mt * 16 + m) * 392 + qoff;
#pragma unroll
          for (int e = 0; e < 16; ++e) {
            int v  = e >> 1;
            int kk = (v < 4) ? (kb + (v << 1) + (e & 1))
                             : (16 + kb + ((v - 4) << 1) + (e & 1));
            a[e] = (kk < 16) ? row[kk] : (_Float16)0.0f;
          }
          int n = lane & 15, kb2 = (lane & 16) ? 8 : 0;
          const _Float16* krow = sQKV + (nt * 16 + n) * 392 + koff;
#pragma unroll
          for (int e = 0; e < 16; ++e) {
            int kk = kb2 + (e & 7) + ((e >> 3) << 4);
            b[e] = (kk < 16) ? krow[kk] : (_Float16)0.0f;
          }
        }
        v8f acc = zerov();
        acc = wmma16(a, b, acc);
        int hi = (lane >> 4) << 3;
#pragma unroll
        for (int r = 0; r < 8; ++r)
          sSim[(w * 64 + mt * 16 + hi + r) * 68 + nt * 16 + (lane & 15)] = acc[r];
      }
  }
  __syncthreads();

  // softmax + relative-position bias; p stored as f16 in-place (half stride 136)
  for (int rid = tid; rid < 512; rid += 256) {
    int hh = rid >> 6, i = rid & 63;
    int ir = i >> 3, ic = i & 7;
    float* srow = sSim + rid * 68;
    float mx = -1e30f;
    for (int j = 0; j < 64; ++j) {
      int jr = j >> 3, jc = j & 7;
      int ridx = (ir - jr + 7) * 15 + (ic - jc + 7);
      float s = srow[j] + rel_table[ridx * 8 + hh];
      srow[j] = s;
      mx = fmaxf(mx, s);
    }
    float sum = 0.f;
    for (int j = 0; j < 64; ++j) { float e = __expf(srow[j] - mx); sum += e; srow[j] = e; }
    float inv = 1.0f / sum;
    _Float16* prow = (_Float16*)srow;
    for (int j = 0; j < 64; ++j) prow[j] = (_Float16)(srow[j] * inv);  // safe in-place
  }
  __syncthreads();

  // o_h(64x16) = p(64x64) @ v_h(64x16); write into sXq (reused)
  {
    const _Float16* pb = (const _Float16*)sSim + w * 64 * 136;  // half stride 136
    int voff = 256 + w * 16;
    for (int mt = 0; mt < 4; ++mt) {
      v8f acc = zerov();
      for (int kt = 0; kt < 2; ++kt) {
        v16h a = load_a_frag(pb + mt * 16 * 136 + kt * 32, 136, lane);
        v16h b = load_b_frag(sQKV + (kt * 32) * 392 + voff, 392, lane);
        acc = wmma16(a, b, acc);
      }
      int hi = (lane >> 4) << 3;
#pragma unroll
      for (int r = 0; r < 8; ++r)
        sXq[(mt * 16 + hi + r) * 136 + w * 16 + (lane & 15)] = (_Float16)acc[r];
    }
  }
  __syncthreads();

  // GEMM3: out(64x128) = o @ w_out, write fp32 channels [0,128)
  {
    int n0 = w * 16;
    for (int mt = 0; mt < 4; ++mt) {
      v8f acc = zerov();
      for (int kt = 0; kt < 4; ++kt) {
        v16h a = load_a_frag(sXq + mt * 16 * 136 + kt * 32, 136, lane);
        v16h b = load_b_frag(wout16 + kt * 32 * 128 + n0, 128, lane);
        acc = wmma16(a, b, acc);
      }
      int hi = (lane >> 4) << 3;
#pragma unroll
      for (int r = 0; r < 8; ++r) {
        int m = mt * 16 + hi + r;
        out[(size_t)wi * 16384 + m * 256 + n0 + (lane & 15)] = acc[r];
      }
    }
  }
}

// ---------------------------------------------------------------------------
// Kernel 2: iq = img @ q2_w + q2_b  (73728x256 @ 256x128), f16 result.
// Rows of A are contiguous in x16 (window-blocked pixel order); async staged.
// ---------------------------------------------------------------------------
__global__ void __launch_bounds__(256)
q2_proj_kernel(const _Float16* __restrict__ x16, const _Float16* __restrict__ q2w16,
               const float* __restrict__ q2_b, _Float16* __restrict__ iq16) {
  __shared__ _Float16 sA[64 * 264];
  const int tid = threadIdx.x, lane = tid & 31, w = tid >> 5;
  const size_t rb = (size_t)blockIdx.x * 64;
  {
    const char* src = (const char*)(x16 + rb * 256);
    for (int i = tid; i < 2048; i += 256) {
      int row = i >> 5, ch = i & 31;
      async_ld_b128((char*)sA + row * 528 + ch * 16, src + row * 512 + ch * 16);
    }
    wait_async0();
  }
  __syncthreads();
  int n0 = w * 16;
  float bias = q2_b[n0 + (lane & 15)];
  for (int mt = 0; mt < 4; ++mt) {
    v8f acc = zerov();
    for (int kt = 0; kt < 8; ++kt) {
      v16h a = load_a_frag(sA + mt * 16 * 264 + kt * 32, 264, lane);
      v16h b = load_b_frag(q2w16 + kt * 32 * 128 + n0, 128, lane);
      acc = wmma16(a, b, acc);
    }
    int hi = (lane >> 4) << 3;
#pragma unroll
    for (int r = 0; r < 8; ++r) {
      size_t row = rb + mt * 16 + hi + r;
      iq16[row * 128 + n0 + (lane & 15)] = (_Float16)(acc[r] + bias);
    }
  }
}

// ---------------------------------------------------------------------------
// Kernel 3: cd conv 3x3/s8/p1 on iq -> cd_out (B,128,24,24) f32 (small, scalar)
// ---------------------------------------------------------------------------
__global__ void __launch_bounds__(128)
conv_cd_kernel(const _Float16* __restrict__ iq16, const float* __restrict__ cd_w,
               const float* __restrict__ cd_b, float* __restrict__ cd_out) {
  __shared__ _Float16 patch[9 * 128];
  int blk = blockIdx.x;
  int b = blk / 576, rem = blk % 576, y = rem / 24, xx = rem % 24;
  for (int i = threadIdx.x; i < 9 * 128; i += 128) {
    int t = i >> 7, ci = i & 127;
    int ky = t / 3, kx = t % 3;
    int py = y * 8 + ky - 1, px = xx * 8 + kx - 1;
    _Float16 v = (_Float16)0.f;
    if (py >= 0 && py < 192 && px >= 0 && px < 192) {
      int p = ((b * 24 + (py >> 3)) * 24 + (px >> 3)) * 64 + (py & 7) * 8 + (px & 7);
      v = iq16[(size_t)p * 128 + ci];
    }
    patch[i] = v;
  }
  __syncthreads();
  int co = threadIdx.x;
  float acc = cd_b[co];
  for (int t = 0; t < 9; ++t)
    for (int ci = 0; ci < 128; ++ci)
      acc += (float)patch[t * 128 + ci] * cd_w[(co * 128 + ci) * 9 + t];
  cd_out[(size_t)(b * 128 + co) * 576 + y * 24 + xx] = acc;
}

// ---------------------------------------------------------------------------
// Kernel 4: sr conv 3x3/s16/p1 + LayerNorm(channel) + exact GELU -> x1g f16
// one block per (b, position) of the 12x12 grid; keeps f32 inputs for LN fidelity
// ---------------------------------------------------------------------------
__global__ void __launch_bounds__(256)
sr_ln_kernel(const float* __restrict__ x, const float* __restrict__ sr_w,
             const float* __restrict__ sr_b, const float* __restrict__ ln_g,
             const float* __restrict__ ln_b, _Float16* __restrict__ x1g) {
  __shared__ float patch[9 * 256];
  __shared__ float red[256];
  __shared__ float stat[2];
  int blk = blockIdx.x;
  int b = blk / 144, pos = blk % 144, y = pos / 12, xx = pos % 12;
  for (int i = threadIdx.x; i < 9 * 256; i += 256) {
    int t = i >> 8, ci = i & 255;
    int ky = t / 3, kx = t % 3;
    int py = y * 16 + ky - 1, px = xx * 16 + kx - 1;
    float v = 0.f;
    if (py >= 0 && py < 192 && px >= 0 && px < 192) {
      int p = ((b * 24 + (py >> 3)) * 24 + (px >> 3)) * 64 + (py & 7) * 8 + (px & 7);
      v = x[(size_t)p * 256 + ci];
    }
    patch[i] = v;
  }
  __syncthreads();
  int co = threadIdx.x;
  float acc = sr_b[co];
  for (int t = 0; t < 9; ++t)
    for (int ci = 0; ci < 256; ++ci)
      acc += patch[t * 256 + ci] * sr_w[(co * 256 + ci) * 9 + t];
  red[co] = acc;
  __syncthreads();
  for (int s = 128; s > 0; s >>= 1) { if (co < s) red[co] += red[co + s]; __syncthreads(); }
  if (co == 0) stat[0] = red[0] * (1.f / 256.f);
  __syncthreads();
  float mu = stat[0], d = acc - mu;
  red[co] = d * d;
  __syncthreads();
  for (int s = 128; s > 0; s >>= 1) { if (co < s) red[co] += red[co + s]; __syncthreads(); }
  if (co == 0) stat[1] = red[0] * (1.f / 256.f);
  __syncthreads();
  float g  = d * rsqrtf(stat[1] + 1e-5f) * ln_g[co] + ln_b[co];
  float ge = 0.5f * g * (1.f + erff(g * 0.70710678118654752f));
  x1g[(size_t)(b * 144 + pos) * 256 + co] = (_Float16)ge;
}

// ---------------------------------------------------------------------------
// Kernel 5: kv = x1g(288x256) @ kv1_w(256x256) + b -> k1/v1 (B,4,144,32) f16
// one wave per 16x16 tile (36 blocks x 8 waves = 288 tiles)
// ---------------------------------------------------------------------------
__global__ void __launch_bounds__(256)
kv_gemm_kernel(const _Float16* __restrict__ x1g, const _Float16* __restrict__ kv1w16,
               const float* __restrict__ kv1_b, _Float16* __restrict__ k1buf,
               _Float16* __restrict__ v1buf) {
  int lane = threadIdx.x & 31, w = threadIdx.x >> 5;
  int gid = blockIdx.x * 8 + w;        // 0..287
  int mt = gid >> 4, nt = gid & 15;
  int n0 = nt * 16;
  v8f acc = zerov();
  for (int kt = 0; kt < 8; ++kt) {
    v16h a = load_a_frag(x1g + (size_t)(mt * 16) * 256 + kt * 32, 256, lane);
    v16h b = load_b_frag(kv1w16 + kt * 32 * 256 + n0, 256, lane);
    acc = wmma16(a, b, acc);
  }
  float bias = kv1_b[n0 + (lane & 15)];
  int hi = (lane >> 4) << 3;
#pragma unroll
  for (int r = 0; r < 8; ++r) {
    int row = mt * 16 + hi + r;        // 0..287
    int b_ = row / 144, npos = row % 144;
    int j = n0 + (lane & 15);
    float v = acc[r] + bias;
    if (j < 128) {
      int g = j >> 5, dd = j & 31;
      k1buf[((b_ * 4 + g) * 144 + npos) * 32 + dd] = (_Float16)v;
    } else {
      int j2 = j - 128, g = j2 >> 5, dd = j2 & 31;
      v1buf[((b_ * 4 + g) * 144 + npos) * 32 + dd] = (_Float16)v;
    }
  }
}

// ---------------------------------------------------------------------------
// Kernel 6: global attention. block = (b, head g, 64-row tile of 576).
// q gathered from cd_out with the reference's "faithful literal reshape".
// LDS: kT/vT 160x32 f16 (zero-padded), qt 64x40 f16, simb 64x168 f32
// ---------------------------------------------------------------------------
__global__ void __launch_bounds__(256)
global_attn_kernel(const float* __restrict__ cd_out, const _Float16* __restrict__ k1buf,
                   const _Float16* __restrict__ v1buf, float* __restrict__ o1sp) {
  extern __shared__ char smem[];
  _Float16* kT = (_Float16*)smem;                    // 160x32
  _Float16* vT = kT + 160 * 32;                      // 160x32
  _Float16* qt = vT + 160 * 32;                      // 64x40
  float* simb  = (float*)(smem + 160 * 32 * 4 + 64 * 40 * 2);  // 64x168
  int tid = threadIdx.x, lane = tid & 31, w = tid >> 5;
  int blk = blockIdx.x;
  int b = blk / 36, r2 = blk % 36, g = r2 / 9, bn = r2 % 9;

  for (int i = tid; i < 160 * 32; i += 256) {
    int row = i >> 5, dd = i & 31;
    _Float16 kk = (_Float16)0.f, vv = (_Float16)0.f;
    if (row < 144) {
      kk = k1buf[((b * 4 + g) * 144 + row) * 32 + dd];
      vv = v1buf[((b * 4 + g) * 144 + row) * 32 + dd];
    }
    kT[i] = kk; vT[i] = vv;
  }
  for (int i = tid; i < 64 * 32; i += 256) {
    int m = i >> 5, dd = i & 31;
    int n = bn * 64 + m;
    int flat = n * 128 + g * 32 + dd;       // literal reshape scramble
    int c = flat / 576, rem = flat % 576;
    qt[m * 40 + dd] = (_Float16)cd_out[(size_t)(b * 128 + c) * 576 + rem];
  }
  __syncthreads();

  // sim(64x160) = q(64x32) @ k^T ; cols >=144 are exact zeros (masked later)
  for (int t = w; t < 40; t += 8) {
    int mt = t / 10, nt = t % 10;
    v16h a = load_a_frag(qt + mt * 16 * 40, 40, lane);
    v16h bb;
    {
      int n = lane & 15, kb = (lane & 16) ? 8 : 0;
      const _Float16* krow = kT + (nt * 16 + n) * 32;
#pragma unroll
      for (int e = 0; e < 16; ++e) bb[e] = krow[kb + (e & 7) + ((e >> 3) << 4)];
    }
    v8f acc = zerov();
    acc = wmma16(a, bb, acc);
    int hi = (lane >> 4) << 3;
#pragma unroll
    for (int r = 0; r < 8; ++r)
      simb[(mt * 16 + hi + r) * 168 + nt * 16 + (lane & 15)] = acc[r];
  }
  __syncthreads();

  // softmax over 144 valid cols; p as f16 in-place (half stride 336)
  if (tid < 64) {
    float* srow = simb + tid * 168;
    float mx = -1e30f;
    for (int j = 0; j < 144; ++j) { float s = srow[j] * SCALE; srow[j] = s; mx = fmaxf(mx, s); }
    float sum = 0.f;
    for (int j = 0; j < 144; ++j) { float e = __expf(srow[j] - mx); sum += e; srow[j] = e; }
    float inv = 1.f / sum;
    _Float16* prow = (_Float16*)srow;
    for (int j = 0; j < 144; ++j) prow[j] = (_Float16)(srow[j] * inv);
    for (int j = 144; j < 160; ++j) prow[j] = (_Float16)0.f;
  }
  __syncthreads();

  // o(64x32) = p(64x160) @ v(160x32); wave -> (mt = w/2, nt = w&1)
  {
    const _Float16* pb = (const _Float16*)simb;     // half stride 336
    int mt = w >> 1, nt = w & 1;
    v8f acc = zerov();
    for (int kt = 0; kt < 5; ++kt) {
      v16h a = load_a_frag(pb + mt * 16 * 336 + kt * 32, 336, lane);
      v16h bb = load_b_frag(vT + kt * 32 * 32 + nt * 16, 32, lane);
      acc = wmma16(a, bb, acc);
    }
    int hi = (lane >> 4) << 3;
#pragma unroll
    for (int r = 0; r < 8; ++r) {
      int m = mt * 16 + hi + r;
      int npos = bn * 64 + m;
      int c = g * 32 + nt * 16 + (lane & 15);
      o1sp[(size_t)(b * 128 + c) * 576 + npos] = acc[r];
    }
  }
}

// ---------------------------------------------------------------------------
// Kernel 7: x8 nearest upsample + 3x3/s1/p1 conv (cu) as WMMA implicit GEMM.
// block = one 8x8 output cell; taps select one of a 3x3 cell neighborhood.
// C(64x128) = sum_{tap=0..8} A_tap(64x128) @ W_tap(128x128)
// ---------------------------------------------------------------------------
__global__ void __launch_bounds__(256)
conv_cu_kernel(const float* __restrict__ o1sp, const _Float16* __restrict__ cu16,
               const float* __restrict__ cu_b, float* __restrict__ out) {
  __shared__ _Float16 cells[9 * 136];
  int tid = threadIdx.x, lane = tid & 31, w = tid >> 5;
  int blk = blockIdx.x;                      // == window index
  int b = blk / 576, r2 = blk % 576, cy = r2 / 24, cx = r2 % 24;
  for (int i = tid; i < 9 * 128; i += 256) {
    int i9 = i >> 7, ci = i & 127;
    int cy2 = cy + i9 / 3 - 1, cx2 = cx + i9 % 3 - 1;
    _Float16 v = (_Float16)0.f;
    if (cy2 >= 0 && cy2 < 24 && cx2 >= 0 && cx2 < 24)
      v = (_Float16)o1sp[(size_t)(b * 128 + ci) * 576 + cy2 * 24 + cx2];
    cells[i9 * 136 + ci] = v;
  }
  __syncthreads();

  int n0 = w * 16;
  v8f acc[4];
#pragma unroll
  for (int mt = 0; mt < 4; ++mt) acc[mt] = zerov();

  for (int tap = 0; tap < 9; ++tap) {
    int ky = tap / 3, kx = tap % 3;
    for (int kt = 0; kt < 4; ++kt) {
      v16h bb = load_b_frag(cu16 + (size_t)(tap * 128 + kt * 32) * 128 + n0, 128, lane);
#pragma unroll
      for (int mt = 0; mt < 4; ++mt) {
        int m = mt * 16 + (lane & 15);
        int r = m >> 3, s = m & 7;
        int i9 = ((r + ky - 1 + 8) >> 3) * 3 + ((s + kx - 1 + 8) >> 3);
        int kb = (lane & 16) ? 8 : 0;
        const _Float16* row = cells + i9 * 136 + kt * 32;
        v16h a;
#pragma unroll
        for (int e = 0; e < 16; ++e) {
          int v_ = e >> 1;
          int kk = (v_ < 4) ? (kb + (v_ << 1) + (e & 1))
                            : (16 + kb + ((v_ - 4) << 1) + (e & 1));
          a[e] = row[kk];
        }
        acc[mt] = wmma16(a, bb, acc[mt]);
      }
    }
  }
  float bias = cu_b[n0 + (lane & 15)];
  int hi = (lane >> 4) << 3;
#pragma unroll
  for (int mt = 0; mt < 4; ++mt)
#pragma unroll
    for (int r = 0; r < 8; ++r) {
      int m = mt * 16 + hi + r;
      out[(size_t)blk * 16384 + m * 256 + 128 + n0 + (lane & 15)] = acc[mt][r] + bias;
    }
}

// ---------------------------------------------------------------------------
// Host launcher
// ---------------------------------------------------------------------------
extern "C" void kernel_launch(void* const* d_in, const int* in_sizes, int n_in,
                              void* d_out, int out_size, void* d_ws, size_t ws_size,
                              hipStream_t stream) {
  (void)in_sizes; (void)n_in; (void)out_size; (void)ws_size;
  const float* x         = (const float*)d_in[0];
  const float* w_qkv     = (const float*)d_in[1];
  const float* w_out     = (const float*)d_in[2];
  const float* rel_table = (const float*)d_in[3];
  const float* sr_w      = (const float*)d_in[4];
  const float* sr_b      = (const float*)d_in[5];
  const float* ln_g      = (const float*)d_in[6];
  const float* ln_b      = (const float*)d_in[7];
  const float* kv1_w     = (const float*)d_in[8];
  const float* kv1_b     = (const float*)d_in[9];
  const float* q1_w      = (const float*)d_in[10];
  const float* q1_b      = (const float*)d_in[11];
  const float* q2_w      = (const float*)d_in[12];
  const float* q2_b      = (const float*)d_in[13];
  const float* cd_w      = (const float*)d_in[14];
  const float* cd_b      = (const float*)d_in[15];
  const float* cu_w      = (const float*)d_in[16];
  const float* cu_b      = (const float*)d_in[17];
  float* out = (float*)d_out;
  char* ws = (char*)d_ws;

  _Float16* q1w16  = (_Float16*)(ws + 0);         // 32768 f16
  _Float16* wqkv16 = (_Float16*)(ws + 65536);     // 49152 f16
  _Float16* wout16 = (_Float16*)(ws + 163840);    // 16384 f16
  _Float16* q2w16  = (_Float16*)(ws + 196608);    // 32768 f16
  _Float16* kv1w16 = (_Float16*)(ws + 262144);    // 65536 f16
  _Float16* cu16   = (_Float16*)(ws + 393216);    // 147456 f16
  _Float16* iq16   = (_Float16*)(ws + 688128);    // 73728*128 f16
  float*    cd_out = (float*)(ws + 19562496);     // 2*128*576 f32
  _Float16* x1g    = (_Float16*)(ws + 20152320);  // 288*256 f16
  _Float16* k1buf  = (_Float16*)(ws + 20299776);  // 2*4*144*32 f16
  _Float16* v1buf  = (_Float16*)(ws + 20373504);  // 2*4*144*32 f16
  float*    o1sp   = (float*)(ws + 20447232);     // 2*128*576 f32
  _Float16* x16    = (_Float16*)(ws + 21037056);  // 18874368 f16 (input in f16)

  cvt_f16_kernel<<<(32768 + 255) / 256, 256, 0, stream>>>(q1_w, q1w16, 32768);
  cvt_f16_kernel<<<(49152 + 255) / 256, 256, 0, stream>>>(w_qkv, wqkv16, 49152);
  cvt_f16_kernel<<<(16384 + 255) / 256, 256, 0, stream>>>(w_out, wout16, 16384);
  cvt_f16_kernel<<<(32768 + 255) / 256, 256, 0, stream>>>(q2_w, q2w16, 32768);
  cvt_f16_kernel<<<(65536 + 255) / 256, 256, 0, stream>>>(kv1_w, kv1w16, 65536);
  cvt_cu_kernel<<<(147456 + 255) / 256, 256, 0, stream>>>(cu_w, cu16);
  cvt_f16_kernel<<<(18874368 + 255) / 256, 256, 0, stream>>>(x, x16, 18874368);

  // window branch (channels 0..127 of output)
  win_attn_kernel<<<1152, 256, 240640, stream>>>(x16, q1w16, q1_b, wqkv16, wout16,
                                                 rel_table, out);
  // global branch
  q2_proj_kernel<<<1152, 256, 0, stream>>>(x16, q2w16, q2_b, iq16);
  conv_cd_kernel<<<1152, 128, 0, stream>>>(iq16, cd_w, cd_b, cd_out);
  sr_ln_kernel<<<288, 256, 0, stream>>>(x, sr_w, sr_b, ln_g, ln_b, x1g);
  kv_gemm_kernel<<<36, 256, 0, stream>>>(x1g, kv1w16, kv1_b, k1buf, v1buf);
  global_attn_kernel<<<72, 256, 68608, stream>>>(cd_out, k1buf, v1buf, o1sp);
  conv_cu_kernel<<<1152, 256, 0, stream>>>(o1sp, cu16, cu_b, out);
}
